// E_GCL_AT_23295902614269
// MI455X (gfx1250) — matrix operand order repaired and checked by reading, not compiled, not run on hardware
//
#include <hip/hip_runtime.h>
#include <hip/hip_bf16.h>
#include <math.h>

typedef __bf16 bf16_t;
typedef __attribute__((ext_vector_type(16))) __bf16 v16bf;
typedef __attribute__((ext_vector_type(8)))  __bf16 v8bf;
typedef __attribute__((ext_vector_type(8)))  float  v8f;

constexpr int kB = 2048;
constexpr int kT = 64;
constexpr int kD = 128;
constexpr int kC = 3;
constexpr int kF = 6;

// ---------------- WMMA helpers (layouts per CDNA5 ISA 7.12.2) ----------------

__device__ __forceinline__ v8f wmma_bf16(v16bf a, v16bf b, v8f c) {
  return __builtin_amdgcn_wmma_f32_16x16x32_bf16(
      /*neg_a=*/false, a, /*neg_b=*/false, b,
      /*c_mod=*/(short)0, c, /*reuse_a=*/false, /*reuse_b=*/false);
}

// A operand: 16x32 tile of row-major bf16 matrix X (leading dim ld).
// lane L: M = L%16; element e: K = (e&7) + 8*(2*(e>>3) + L/16)
__device__ __forceinline__ v16bf ldA_vec(const bf16_t* X, int ld, int m0, int k0, int lane) {
  int M = lane & 15, hh = lane >> 4;
  const bf16_t* p = X + (m0 + M) * ld + k0 + 8 * hh;
  v8bf lo = *(const v8bf*)(p);
  v8bf hi = *(const v8bf*)(p + 16);
  return __builtin_shufflevector(lo, hi, 0, 1, 2, 3, 4, 5, 6, 7,
                                 8, 9, 10, 11, 12, 13, 14, 15);
}

// B operand from an [N][K]-major buffer (Xt[n*ld + k]): B[K][N] = Xt[n0+N][k0+K].
__device__ __forceinline__ v16bf ldB_nk(const bf16_t* Xt, int ld, int k0, int n0, int lane) {
  int N = lane & 15, hh = lane >> 4;
  const bf16_t* p = Xt + (n0 + N) * ld + k0 + 8 * hh;
  v8bf lo = *(const v8bf*)(p);
  v8bf hi = *(const v8bf*)(p + 16);
  return __builtin_shufflevector(lo, hi, 0, 1, 2, 3, 4, 5, 6, 7,
                                 8, 9, 10, 11, 12, 13, 14, 15);
}

// 64x128 = relu( X(64x128) @ W(128x128) + bias ).  Wt is W transposed in LDS.
// 8 waves; wave w owns cols [16w,16w+16). Output modes resolved at compile time.
template <bool OUT_RM, bool OUT_TR>
__device__ __forceinline__ void layer128(const bf16_t* Xin, const bf16_t* Wt,
                                         const float* biasLds, bf16_t* ldsOut,
                                         bf16_t* ldsOutT, int lane, int w) {
  int n0 = w * 16;
  int N = lane & 15, hh = lane >> 4;
  float bias = biasLds[n0 + N];
  #pragma unroll
  for (int mt = 0; mt < 4; ++mt) {
    v8f acc = {};
    #pragma unroll
    for (int kt = 0; kt < 4; ++kt)
      acc = wmma_bf16(ldA_vec(Xin, kD, mt * 16, kt * 32, lane),
                      ldB_nk(Wt, kD, kt * 32, n0, lane), acc);
    #pragma unroll
    for (int r = 0; r < 8; ++r) {
      int row = mt * 16 + r + 8 * hh;
      float val = fmaxf(acc[r] + bias, 0.f);
      bf16_t bv = (bf16_t)val;
      if constexpr (OUT_RM) ldsOut[row * kD + n0 + N] = bv;
      if constexpr (OUT_TR) ldsOutT[(n0 + N) * kT + row] = bv;
    }
  }
}

// Stage W (row-major DxD f32) transposed into LDS bf16: sWt[n*kD+k] = W[k*kD+n]
__device__ __forceinline__ void stageWT(const float* __restrict__ W, bf16_t* sWt,
                                        const float* __restrict__ bvec, float* sB,
                                        int tid) {
  for (int i = tid; i < kD * kD; i += 256) {
    int k = i >> 7, n = i & 127;
    sWt[n * kD + k] = (bf16_t)W[i];
  }
  if (tid < kD) sB[tid] = bvec[tid];
}

// ---------------- Kernel A: fused qkv MLPs + cm_v + attention ----------------

__global__ __launch_bounds__(256) void k_fused(
    const float* __restrict__ h, const float* __restrict__ coord,
    const float* __restrict__ Wq1, const float* __restrict__ bq1,
    const float* __restrict__ Wq2, const float* __restrict__ bq2,
    const float* __restrict__ Wk1, const float* __restrict__ bk1,
    const float* __restrict__ Wk2, const float* __restrict__ bk2,
    const float* __restrict__ Wv1, const float* __restrict__ bv1,
    const float* __restrict__ Wv2, const float* __restrict__ bv2,
    const float* __restrict__ Wc1, const float* __restrict__ bc1,
    const float* __restrict__ Wc2,
    float* __restrict__ out1, float* __restrict__ xagg) {
  __shared__ alignas(16) bf16_t sA[kT * kD];    // h (bf16), A operand
  __shared__ alignas(16) bf16_t sH[kT * kD];    // MLP hidden
  __shared__ alignas(16) bf16_t sQ[kT * kD];    // q rows (A for scores)
  __shared__ alignas(16) bf16_t sK[kT * kD];    // k rows ([N][K] B for scores)
  __shared__ alignas(16) bf16_t sV[kT * kD];    // v rows (A for cm GEMM)
  __shared__ alignas(16) bf16_t sVt[kD * kT];   // v transposed ([N][K] B for h_agg)
  __shared__ alignas(16) bf16_t sWt[kD * kD];   // current weight, transposed
  __shared__ float  sS[kT * kT];                // scores / exp
  __shared__ alignas(16) bf16_t sAl[kT * kT];   // alpha bf16 (A for h_agg)
  __shared__ float  sB[kD];
  __shared__ float  sCm[kT];
  __shared__ float  sCo[kT * 3];

  int b = blockIdx.x;
  int tid = threadIdx.x, lane = tid & 31, w = tid >> 5;
  size_t base = (size_t)b * kT * kD;
  const float* hb = h + base;
  for (int i = tid; i < kT * kD; i += 256) sA[i] = (bf16_t)hb[i];
  if (tid < kT * 3) sCo[tid] = coord[(size_t)b * kT * 3 + tid];
  __syncthreads();

  // ---- q = relu(relu(h@Wq1+bq1)@Wq2+bq2) ----
  stageWT(Wq1, sWt, bq1, sB, tid); __syncthreads();
  layer128<true, false>(sA, sWt, sB, sH, nullptr, lane, w); __syncthreads();
  stageWT(Wq2, sWt, bq2, sB, tid); __syncthreads();
  layer128<true, false>(sH, sWt, sB, sQ, nullptr, lane, w); __syncthreads();
  // ---- k ----
  stageWT(Wk1, sWt, bk1, sB, tid); __syncthreads();
  layer128<true, false>(sA, sWt, sB, sH, nullptr, lane, w); __syncthreads();
  stageWT(Wk2, sWt, bk2, sB, tid); __syncthreads();
  layer128<true, false>(sH, sWt, sB, sK, nullptr, lane, w); __syncthreads();
  // ---- v (row-major + transposed) ----
  stageWT(Wv1, sWt, bv1, sB, tid); __syncthreads();
  layer128<true, false>(sA, sWt, sB, sH, nullptr, lane, w); __syncthreads();
  stageWT(Wv2, sWt, bv2, sB, tid); __syncthreads();
  layer128<true, true>(sH, sWt, sB, sV, sVt, lane, w); __syncthreads();
  // ---- cm_v = relu(v@Wc1+bc1) @ Wc2 ----
  stageWT(Wc1, sWt, bc1, sB, tid); __syncthreads();
  layer128<true, false>(sV, sWt, sB, sH, nullptr, lane, w); __syncthreads();
  if (tid < kT) {
    float s = 0.f;
    for (int d = 0; d < kD; ++d) s += (float)sH[tid * kD + d] * Wc2[d];
    sCm[tid] = s;
  }
  __syncthreads();

  // ---- scores = q @ k^T : 4x4 tiles, 2 per wave, K=128 ----
  {
    int N = lane & 15, hh = lane >> 4;
    #pragma unroll
    for (int tt = 0; tt < 2; ++tt) {
      int tile = w + tt * 8;
      int mt = tile >> 2, nt = tile & 3;
      v8f acc = {};
      #pragma unroll
      for (int kt = 0; kt < 4; ++kt)
        acc = wmma_bf16(ldA_vec(sQ, kD, mt * 16, kt * 32, lane),
                        ldB_nk(sK, kD, kt * 32, nt * 16, lane), acc);
      #pragma unroll
      for (int r = 0; r < 8; ++r)
        sS[(mt * 16 + r + 8 * hh) * kT + nt * 16 + N] = acc[r];
    }
  }
  __syncthreads();

  // ---- softmax rows + fused x_agg (one thread per query row) ----
  if (tid < kT) {
    int i = tid;
    float mx = -1e30f;
    for (int j = 0; j < kT; ++j) mx = fmaxf(mx, sS[i * kT + j]);
    float sum = 0.f;
    for (int j = 0; j < kT; ++j) {
      float e = __expf(sS[i * kT + j] - mx);
      sS[i * kT + j] = e;
      sum += e;
    }
    float inv = 1.f / sum;
    float s0 = 0.f, s1 = 0.f, s2 = 0.f, sw = 0.f;
    for (int j = 0; j < kT; ++j) {
      float a = sS[i * kT + j] * inv;
      sAl[i * kT + j] = (bf16_t)a;
      float wj = a * sCm[j];
      sw += wj;
      s0 += wj * sCo[j * 3 + 0];
      s1 += wj * sCo[j * 3 + 1];
      s2 += wj * sCo[j * 3 + 2];
    }
    float* xo = xagg + ((size_t)b * kT + i) * 3;
    xo[0] = s0 - sw * sCo[i * 3 + 0];
    xo[1] = s1 - sw * sCo[i * 3 + 1];
    xo[2] = s2 - sw * sCo[i * 3 + 2];
  }
  __syncthreads();

  // ---- h_agg = alpha @ v ; out1 = h + h_agg ----
  {
    int n0 = w * 16;
    int N = lane & 15, hh = lane >> 4;
    #pragma unroll
    for (int mt = 0; mt < 4; ++mt) {
      v8f acc = {};
      #pragma unroll
      for (int kt = 0; kt < 2; ++kt)
        acc = wmma_bf16(ldA_vec(sAl, kT, mt * 16, kt * 32, lane),
                        ldB_nk(sVt, kT, kt * 32, n0, lane), acc);
      #pragma unroll
      for (int r = 0; r < 8; ++r) {
        int row = mt * 16 + r + 8 * hh;
        size_t idx = base + (size_t)row * kD + n0 + N;
        out1[idx] = h[idx] + acc[r];
      }
    }
  }
}

// ---------------- Kernel B: fused spectral pipeline --------------------------

__device__ __forceinline__ void smlp6(const float* x, const float* __restrict__ W1,
                                      const float* __restrict__ W2, float* out) {
  float hbuf[16];
  #pragma unroll
  for (int u = 0; u < 16; ++u) {
    float s = 0.f;
    #pragma unroll
    for (int f = 0; f < kF; ++f) s += x[f] * W1[f * 16 + u];
    hbuf[u] = fmaxf(s, 0.f);
  }
  #pragma unroll
  for (int f = 0; f < kF; ++f) {
    float s = 0.f;
    #pragma unroll
    for (int u = 0; u < 16; ++u) s += hbuf[u] * W2[u * kF + f];
    out[f] = fmaxf(s, 0.f);
  }
}

__global__ __launch_bounds__(64) void k_spec(
    const float* __restrict__ stft, const float* __restrict__ coord,
    const float* __restrict__ Wqs1, const float* __restrict__ Wqs2,
    const float* __restrict__ Wks1, const float* __restrict__ Wks2,
    const float* __restrict__ Wvs1, const float* __restrict__ Wvs2,
    const float* __restrict__ Wcs1, const float* __restrict__ bcs1,
    const float* __restrict__ Wcs2, const float* __restrict__ bcs2,
    float* __restrict__ out3, float* __restrict__ mag_out,
    float* __restrict__ xaggs) {
  __shared__ float sQ[kC * kT * kF];
  __shared__ float sK[kC * kT * kF];
  __shared__ float sV[kC * kT * kF];
  __shared__ float sE[kC * kT * kT];   // unnormalized exp(scores - max)
  __shared__ float sInv[kC * kT];      // per-row 1/sum
  __shared__ float sCms[kT];
  __shared__ float sCo[kT * 3];

  int b = blockIdx.x, i = threadIdx.x;   // i = this thread's row t
  size_t sbase = (size_t)b * kC * kT * kF;

  // spectral MLPs for row i, all 3 channels, directly into LDS; stft cached.
  float xs[kC][kF];
  float mag2v[kF] = {};
  for (int c = 0; c < kC; ++c) {
    size_t off = sbase + (size_t)(c * kT + i) * kF;
    float o[kF];
    #pragma unroll
    for (int f = 0; f < kF; ++f) xs[c][f] = stft[off + f];
    smlp6(xs[c], Wqs1, Wqs2, o);
    #pragma unroll
    for (int f = 0; f < kF; ++f) sQ[(c * kT + i) * kF + f] = o[f];
    smlp6(xs[c], Wks1, Wks2, o);
    #pragma unroll
    for (int f = 0; f < kF; ++f) sK[(c * kT + i) * kF + f] = o[f];
    smlp6(xs[c], Wvs1, Wvs2, o);
    #pragma unroll
    for (int f = 0; f < kF; ++f) { sV[(c * kT + i) * kF + f] = o[f]; mag2v[f] += o[f] * o[f]; }
  }
  // cms = relu(vs_mag@Wcs1+bcs1)@Wcs2 + bcs2
  {
    float mg[kF];
    #pragma unroll
    for (int f = 0; f < kF; ++f) mg[f] = sqrtf(mag2v[f]);
    float out = bcs2[0];
    #pragma unroll
    for (int u = 0; u < 16; ++u) {
      float s = bcs1[u];
      #pragma unroll
      for (int f = 0; f < kF; ++f) s += mg[f] * Wcs1[f * 16 + u];
      out += fmaxf(s, 0.f) * Wcs2[u];
    }
    sCms[i] = out;
  }
  for (int t = i; t < kT * 3; t += 64) sCo[t] = coord[(size_t)b * kT * 3 + t];
  __syncthreads();

  // scores + softmax (unnormalized exp + 1/sum)
  for (int c = 0; c < kC; ++c) {
    const float* qrow = &sQ[(c * kT + i) * kF];
    float mx = -1e30f;
    for (int j = 0; j < kT; ++j) {
      const float* krow = &sK[(c * kT + j) * kF];
      float s = 0.f;
      #pragma unroll
      for (int f = 0; f < kF; ++f) s += qrow[f] * krow[f];
      sE[(c * kT + i) * kT + j] = s;
      mx = fmaxf(mx, s);
    }
    float sum = 0.f;
    for (int j = 0; j < kT; ++j) {
      float e = __expf(sE[(c * kT + i) * kT + j] - mx);
      sE[(c * kT + i) * kT + j] = e;
      sum += e;
    }
    sInv[c * kT + i] = 1.f / sum;
  }
  __syncthreads();

  // h_agg_s -> out3 = stft + h_agg_s (stft from registers), mag over C
  float mag2[kF] = {};
  for (int c = 0; c < kC; ++c) {
    float inv = sInv[c * kT + i];
    float acc[kF] = {};
    for (int j = 0; j < kT; ++j) {
      float a = sE[(c * kT + i) * kT + j];
      const float* vrow = &sV[(c * kT + j) * kF];
      #pragma unroll
      for (int f = 0; f < kF; ++f) acc[f] += a * vrow[f];
    }
    size_t off = sbase + (size_t)(c * kT + i) * kF;
    #pragma unroll
    for (int f = 0; f < kF; ++f) {
      float hv = acc[f] * inv;
      out3[off + f] = xs[c][f] + hv;
      mag2[f] += hv * hv;
    }
  }
  size_t mo = ((size_t)b * kT + i) * kF;
  #pragma unroll
  for (int f = 0; f < kF; ++f) mag_out[mo + f] = sqrtf(mag2[f]);

  // x_agg_s with asm[b,i,j] = ||alpha_s[b,:,j,i]|| (transposed norm)
  float s0 = 0.f, s1 = 0.f, s2 = 0.f, sw = 0.f;
  for (int j = 0; j < kT; ++j) {
    float a2 = 0.f;
    #pragma unroll
    for (int c = 0; c < kC; ++c) {
      float a = sE[(c * kT + j) * kT + i] * sInv[c * kT + j];
      a2 += a * a;
    }
    float wj = sqrtf(a2) * sCms[j];
    sw += wj;
    s0 += wj * sCo[j * 3 + 0];
    s1 += wj * sCo[j * 3 + 1];
    s2 += wj * sCo[j * 3 + 2];
  }
  float* xo = xaggs + ((size_t)b * kT + i) * 3;
  xo[0] = s0 - sw * sCo[i * 3 + 0];
  xo[1] = s1 - sw * sCo[i * 3 + 1];
  xo[2] = s2 - sw * sCo[i * 3 + 2];
}

// ---------------- Kernel C: gamma + coord_out --------------------------------

__global__ __launch_bounds__(256) void k_final(
    const float* __restrict__ h, const float* __restrict__ coord,
    const float* __restrict__ mag, const float* __restrict__ xagg,
    const float* __restrict__ xaggs, const float* __restrict__ Wg,
    const float* __restrict__ bg, float* __restrict__ out2) {
  int idx = blockIdx.x * 256 + threadIdx.x;   // (b,t)
  if (idx >= kB * kT) return;
  const float* hr = h + (size_t)idx * kD;
  float s = bg[0];
  for (int d = 0; d < kD; ++d) s += hr[d] * Wg[d];
  const float* mr = mag + (size_t)idx * kF;
  #pragma unroll
  for (int f = 0; f < kF; ++f) s += mr[f] * Wg[kD + f];
  float g = 1.f / (1.f + __expf(-s));
  #pragma unroll
  for (int d = 0; d < 3; ++d) {
    size_t o = (size_t)idx * 3 + d;
    out2[o] = coord[o] + g * xagg[o] + (1.f - g) * xaggs[o];
  }
}

// ---------------- Host launcher ----------------------------------------------

extern "C" void kernel_launch(void* const* d_in, const int* in_sizes, int n_in,
                              void* d_out, int out_size, void* d_ws, size_t ws_size,
                              hipStream_t stream) {
  const float* h     = (const float*)d_in[0];
  const float* coord = (const float*)d_in[1];
  const float* stft  = (const float*)d_in[2];
  const float* Wq1 = (const float*)d_in[3],  *bq1 = (const float*)d_in[4];
  const float* Wq2 = (const float*)d_in[5],  *bq2 = (const float*)d_in[6];
  const float* Wk1 = (const float*)d_in[7],  *bk1 = (const float*)d_in[8];
  const float* Wk2 = (const float*)d_in[9],  *bk2 = (const float*)d_in[10];
  const float* Wv1 = (const float*)d_in[11], *bv1 = (const float*)d_in[12];
  const float* Wv2 = (const float*)d_in[13], *bv2 = (const float*)d_in[14];
  const float* Wqs1 = (const float*)d_in[15], *Wqs2 = (const float*)d_in[16];
  const float* Wks1 = (const float*)d_in[17], *Wks2 = (const float*)d_in[18];
  const float* Wvs1 = (const float*)d_in[19], *Wvs2 = (const float*)d_in[20];
  const float* Wc1 = (const float*)d_in[21], *bc1 = (const float*)d_in[22];
  const float* Wc2 = (const float*)d_in[23];
  const float* Wcs1 = (const float*)d_in[24], *bcs1 = (const float*)d_in[25];
  const float* Wcs2 = (const float*)d_in[26], *bcs2 = (const float*)d_in[27];
  const float* Wg = (const float*)d_in[28], *bg = (const float*)d_in[29];

  const size_t nBTD = (size_t)kB * kT * kD;   // 16,777,216
  const size_t nBT  = (size_t)kB * kT;        // 131,072

  float* magw  = (float*)d_ws;                // (B,T,F)
  float* xagg  = magw + nBT * kF;             // (B,T,3)
  float* xaggs = xagg + nBT * 3;              // (B,T,3)

  float* out1 = (float*)d_out;                // h + h_agg       (B,T,D)
  float* out2 = out1 + nBTD;                  // coord_out       (B,T,3)
  float* out3 = out2 + nBT * 3;               // stft + h_agg_s  (B,C,T,F)

  k_fused<<<kB, 256, 0, stream>>>(h, coord,
      Wq1, bq1, Wq2, bq2, Wk1, bk1, Wk2, bk2, Wv1, bv1, Wv2, bv2,
      Wc1, bc1, Wc2, out1, xagg);

  k_spec<<<kB, 64, 0, stream>>>(stft, coord,
      Wqs1, Wqs2, Wks1, Wks2, Wvs1, Wvs2, Wcs1, bcs1, Wcs2, bcs2,
      out3, magw, xaggs);

  k_final<<<(int)((nBT + 255) / 256), 256, 0, stream>>>(h, coord, magw, xagg,
                                                        xaggs, Wg, bg, out2);
}
